// GConvSE3_32813550141528
// MI455X (gfx1250) — matrix operand order repaired and checked by reading, not compiled, not run on hardware
//
#include <hip/hip_runtime.h>

// ---------------------------------------------------------------------------
// SE(3) TFN graph conv for MI455X (gfx1250, wave32, WMMA).
// Radial-MLP GEMM chain runs on v_wmma_f32_16x16x32_f16 (16 edges = M tile).
// ---------------------------------------------------------------------------

#define N_NODES 20000
#define N_EDGES 320000
#define N_TILES (N_EDGES / 16)   // 20000
#define NW 4                     // waves per block
#define NBLK 250                 // 250*4 = 1000 waves -> exactly 20 tiles/wave

typedef __attribute__((ext_vector_type(16))) _Float16 v16h;
typedef __attribute__((ext_vector_type(8)))  float    v8f;

// ---- WMMA helpers ----------------------------------------------------------

__device__ __forceinline__ v8f wmma16(v16h a, v16h b, v8f c) {
  // D = A(16x32 f16) * B(32x16 f16) + C(16x16 f32)
  return __builtin_amdgcn_wmma_f32_16x16x32_f16(
      /*neg_a=*/false, a, /*neg_b=*/false, b,
      /*c_mod=*/(short)0, c, /*reuse_a=*/false, /*reuse_b=*/false);
}

// A-matrix fragment (16x32, f16) from row-major LDS tile [16][32].
// Lane l<16 : row M=l,    elems 0..7 = K 0..7,  elems 8..15 = K 16..23
// Lane l>=16: row M=l-16, elems 0..7 = K 8..15, elems 8..15 = K 24..31
__device__ __forceinline__ v16h load_A(const _Float16* rowmaj, int lane) {
  const int row = lane & 15;
  const int kb  = (lane < 16) ? 0 : 8;
  const _Float16* p = rowmaj + row * 32 + kb;
  v16h a;
#pragma unroll
  for (int j = 0; j < 8; ++j) a[j] = p[j];
#pragma unroll
  for (int j = 0; j < 8; ++j) a[8 + j] = p[16 + j];
  return a;
}

// B-matrix fragment from pre-swizzled LDS (contiguous 16 halves per lane).
__device__ __forceinline__ v16h load_B(const _Float16* frag, int lane) {
  const _Float16* p = frag + lane * 16;
  v16h b;
#pragma unroll
  for (int j = 0; j < 16; ++j) b[j] = p[j];
  return b;
}

// ---- init / finish kernels -------------------------------------------------

__global__ void se3_zero_kernel(float* __restrict__ out, float* __restrict__ cnt) {
  int i = blockIdx.x * 256 + threadIdx.x;
  if (i < N_NODES * 32) out[i] = 0.0f;
  if (i < N_NODES)      cnt[i] = 0.0f;
}

__global__ void se3_norm_kernel(float* __restrict__ out, const float* __restrict__ cnt) {
  int i = blockIdx.x * 256 + threadIdx.x;
  if (i < N_NODES * 32) out[i] = out[i] / fmaxf(cnt[i >> 5], 1.0f);
}

// ---- main edge kernel ------------------------------------------------------

__global__ __launch_bounds__(NW * 32) void se3_edge_kernel(
    const int*   __restrict__ src,    const int*   __restrict__ dst,
    const float* __restrict__ feat0,  const float* __restrict__ feat1,
    const float* __restrict__ wEmb,   const float* __restrict__ radial,
    const float* __restrict__ b00g,   const float* __restrict__ b01g,
    const float* __restrict__ b10g,   const float* __restrict__ b11g,
    const float* __restrict__ W1,     const float* __restrict__ b1,
    const float* __restrict__ W2,     const float* __restrict__ b2,
    const float* __restrict__ W3_00,  const float* __restrict__ b3_00,
    const float* __restrict__ W3_01,  const float* __restrict__ b3_01,
    const float* __restrict__ W3_10,  const float* __restrict__ b3_10,
    const float* __restrict__ W3_11,  const float* __restrict__ b3_11,
    const float* __restrict__ ws0g,   const float* __restrict__ ws1g,
    float* __restrict__ outAcc,       float* __restrict__ cnt)
{
  // Weights in WMMA-B-fragment layout, f16.
  __shared__ _Float16 sW1[4 * 2 * 512];        //  8 KB
  __shared__ _Float16 sW2[4 * 2 * 512];        //  8 KB
  __shared__ _Float16 sW3[(3 * 4 + 12) * 512]; // 24 KB
  __shared__ float sB1[4 * 32], sB2[4 * 32], sB3[3 * 64 + 192];
  __shared__ float sWS0[64], sWS1[64];
  __shared__ _Float16 sA[NW][16 * 32];         // per-wave A staging (f16)
  __shared__ float    sR[NW][16 * 192];        // per-wave stage-3 output (f32)

  const int tid = threadIdx.x;

  // ---- swizzle weights into B-fragment layout -----------------------------
  for (int idx = tid; idx < 4096; idx += NW * 32) {
    int j = idx & 15, ln = (idx >> 4) & 31, pf = idx >> 9;   // pf = p*2+f
    int p = pf >> 1, f = pf & 1;
    int k = (j & 7) + ((j >> 3) << 4) + ((ln >> 4) << 3);
    int n = f * 16 + (ln & 15);
    sW1[idx] = (_Float16)((k < 17) ? W1[(p * 17 + k) * 32 + n] : 0.0f);
    sW2[idx] = (_Float16)(W2[(p * 32 + k) * 32 + n]);
  }
  for (int idx = tid; idx < 6144; idx += NW * 32) {
    int j = idx & 15, ln = (idx >> 4) & 31, pf = idx >> 9;   // pf = 0..11
    int k = (j & 7) + ((j >> 3) << 4) + ((ln >> 4) << 3);
    // pairs 0..2 : pf = p*4 + f
    {
      int p = pf >> 2, f = pf & 3;
      int n = f * 16 + (ln & 15);
      const float* W3 = (p == 0) ? W3_00 : ((p == 1) ? W3_01 : W3_10);
      sW3[idx] = (_Float16)(W3[k * 64 + n]);
    }
    // pair 3 : pf = f (0..11)
    {
      int n = pf * 16 + (ln & 15);
      sW3[6144 + idx] = (_Float16)(W3_11[k * 192 + n]);
    }
  }
  for (int i = tid; i < 128; i += NW * 32) { sB1[i] = b1[i]; sB2[i] = b2[i]; }
  for (int i = tid; i < 384; i += NW * 32) {
    sB3[i] = (i < 64) ? b3_00[i] : (i < 128) ? b3_01[i - 64]
           : (i < 192) ? b3_10[i - 128] : b3_11[i - 192];
  }
  for (int i = tid; i < 64; i += NW * 32) { sWS0[i] = ws0g[i]; sWS1[i] = ws1g[i]; }
  __syncthreads();

  const int wib   = tid >> 5;
  const int lane  = tid & 31;
  const int gwave = blockIdx.x * NW + wib;  // 0..999
  _Float16* myA = sA[wib];
  float*    myR = sR[wib];
  const int eL   = lane & 15;
  const int hh   = lane >> 4;   // 0: deg-0 outputs, 1: deg-1 outputs
  const int colL = lane & 15;   // column owned in C fragments
  const int mb   = (lane < 16) ? 0 : 8;

  for (int tile = gwave; tile < N_TILES; tile += NBLK * NW) {
    const int e  = tile * 16 + eL;
    const int sN = src[e];
    const int dN = dst[e];

    // prefetch next tile's edge embedding through GL2
    {
      int nt = tile + NBLK * NW;
      if (nt < N_TILES) __builtin_prefetch(wEmb + (size_t)(nt * 16 + eL) * 16, 0, 0);
    }

    // ---- per-edge features / basis / self-interaction (registers) --------
    float fs0[8];
#pragma unroll
    for (int i = 0; i < 8; ++i) fs0[i] = feat0[sN * 8 + i];
    float fs1[24];
#pragma unroll
    for (int i = 0; i < 24; ++i) fs1[i] = feat1[sN * 24 + i];

    float m0[8];     // deg-0 messages (hh==0)
    float m1[24];    // deg-1 messages (hh==1), [o*3+m]
    float g10[8];    // hh==0: sum_n b10[n]*fs1[i,n]
    float b00v = 0.0f, b01v[3] = {0, 0, 0};
    float u[72];     // hh==1: u[(i*3+f)*3+m] = sum_n b11[m,n,f]*fs1[i,n]

    if (hh == 0) {
      b00v = b00g[e];
      float b10v[3];
#pragma unroll
      for (int n = 0; n < 3; ++n) b10v[n] = b10g[e * 3 + n];
#pragma unroll
      for (int i = 0; i < 8; ++i)
        g10[i] = b10v[0] * fs1[i * 3] + b10v[1] * fs1[i * 3 + 1] + b10v[2] * fs1[i * 3 + 2];
      float fd0[8];
#pragma unroll
      for (int i = 0; i < 8; ++i) fd0[i] = feat0[dN * 8 + i];
#pragma unroll
      for (int o = 0; o < 8; ++o) {
        float a = 0.0f;
#pragma unroll
        for (int i = 0; i < 8; ++i) a += sWS0[o * 8 + i] * fd0[i];
        m0[o] = a;
      }
    } else {
#pragma unroll
      for (int m = 0; m < 3; ++m) b01v[m] = b01g[e * 3 + m];
      float b11v[27];
#pragma unroll
      for (int q = 0; q < 27; ++q) b11v[q] = b11g[e * 27 + q];
#pragma unroll
      for (int i = 0; i < 8; ++i)
#pragma unroll
        for (int f = 0; f < 3; ++f)
#pragma unroll
          for (int m = 0; m < 3; ++m)
            u[(i * 3 + f) * 3 + m] = b11v[m * 9 + 0 * 3 + f] * fs1[i * 3 + 0]
                                   + b11v[m * 9 + 1 * 3 + f] * fs1[i * 3 + 1]
                                   + b11v[m * 9 + 2 * 3 + f] * fs1[i * 3 + 2];
      float fd1[24];
#pragma unroll
      for (int i = 0; i < 24; ++i) fd1[i] = feat1[dN * 24 + i];
#pragma unroll
      for (int o = 0; o < 8; ++o)
#pragma unroll
        for (int m = 0; m < 3; ++m) {
          float a = 0.0f;
#pragma unroll
          for (int i = 0; i < 8; ++i) a += sWS1[o * 8 + i] * fd1[i * 3 + m];
          m1[o * 3 + m] = a;
        }
    }

    // ---- build ef tile [16 edges x 32] (K 17..31 zero-padded) -------------
    __syncthreads();
    {
      _Float16* row = myA + eL * 32;
      if (hh == 0) {
#pragma unroll
        for (int k = 0; k < 16; ++k) row[k] = (_Float16)wEmb[e * 16 + k];
      } else {
        row[16] = (_Float16)radial[e];
#pragma unroll
        for (int k = 17; k < 32; ++k) row[k] = (_Float16)0.0f;
      }
    }
    __syncthreads();
    const v16h aEF = load_A(myA, lane);   // kept in regs for all 4 pairs

    // ---- 4 degree-pair MLP chains (all WMMA) ------------------------------
#pragma unroll
    for (int p = 0; p < 4; ++p) {
      const int ncols = (p < 3) ? 64 : 192;
      const int nf    = ncols >> 4;
      const int w3off = (p < 3) ? p * 2048 : 6144;
      const int b3off = (p < 3) ? p * 64 : 192;

      // stage 1: h1 = relu(ef @ W1[p] + b1[p])   (N=32 -> 2 WMMAs)
      {
        float bv0 = sB1[p * 32 + colL];
        float bv1 = sB1[p * 32 + 16 + colL];
        v8f c0, c1;
#pragma unroll
        for (int j = 0; j < 8; ++j) { c0[j] = bv0; c1[j] = bv1; }
        v16h bf0 = load_B(sW1 + (p * 2 + 0) * 512, lane);
        v16h bf1 = load_B(sW1 + (p * 2 + 1) * 512, lane);
        c0 = wmma16(aEF, bf0, c0);
        c1 = wmma16(aEF, bf1, c1);
        __syncthreads();
#pragma unroll
        for (int j = 0; j < 8; ++j) {
          myA[(mb + j) * 32 + colL]      = (_Float16)fmaxf(c0[j], 0.0f);
          myA[(mb + j) * 32 + 16 + colL] = (_Float16)fmaxf(c1[j], 0.0f);
        }
      }
      __syncthreads();
      const v16h aH1 = load_A(myA, lane);
      __syncthreads();

      // stage 2: h2 = relu(h1 @ W2[p] + b2[p])   (N=32 -> 2 WMMAs)
      {
        float bv0 = sB2[p * 32 + colL];
        float bv1 = sB2[p * 32 + 16 + colL];
        v8f c0, c1;
#pragma unroll
        for (int j = 0; j < 8; ++j) { c0[j] = bv0; c1[j] = bv1; }
        v16h bf0 = load_B(sW2 + (p * 2 + 0) * 512, lane);
        v16h bf1 = load_B(sW2 + (p * 2 + 1) * 512, lane);
        c0 = wmma16(aH1, bf0, c0);
        c1 = wmma16(aH1, bf1, c1);
#pragma unroll
        for (int j = 0; j < 8; ++j) {
          myA[(mb + j) * 32 + colL]      = (_Float16)fmaxf(c0[j], 0.0f);
          myA[(mb + j) * 32 + 16 + colL] = (_Float16)fmaxf(c1[j], 0.0f);
        }
      }
      __syncthreads();
      const v16h aH2 = load_A(myA, lane);

      // stage 3: R = h2 @ W3[p] + b3[p]   (nf WMMAs, f32 to LDS)
#pragma unroll
      for (int f = 0; f < nf; ++f) {
        float bv = sB3[b3off + f * 16 + colL];
        v8f c;
#pragma unroll
        for (int j = 0; j < 8; ++j) c[j] = bv;
        v16h bf = load_B(sW3 + w3off + f * 512, lane);
        c = wmma16(aH2, bf, c);
#pragma unroll
        for (int j = 0; j < 8; ++j) myR[(mb + j) * ncols + f * 16 + colL] = c[j];
      }
      __syncthreads();

      // ---- conv contraction for this pair (2 lanes per edge) -------------
      if (p == 0) {
        if (hh == 0) {
          const float* R = myR + eL * 64;
#pragma unroll
          for (int o = 0; o < 8; ++o) {
            float s = 0.0f;
#pragma unroll
            for (int i = 0; i < 8; ++i) s += R[o * 8 + i] * fs0[i];
            m0[o] += b00v * s;
          }
        }
      } else if (p == 1) {
        if (hh == 1) {
          const float* R = myR + eL * 64;
#pragma unroll
          for (int o = 0; o < 8; ++o) {
            float s = 0.0f;
#pragma unroll
            for (int i = 0; i < 8; ++i) s += R[o * 8 + i] * fs0[i];
#pragma unroll
            for (int m = 0; m < 3; ++m) m1[o * 3 + m] += b01v[m] * s;
          }
        }
      } else if (p == 2) {
        if (hh == 0) {
          const float* R = myR + eL * 64;
#pragma unroll
          for (int o = 0; o < 8; ++o) {
            float s = 0.0f;
#pragma unroll
            for (int i = 0; i < 8; ++i) s += R[o * 8 + i] * g10[i];
            m0[o] += s;
          }
        }
      } else {
        if (hh == 1) {
          const float* R = myR + eL * 192;   // col = (o*8+i)*3+f = o*24+i*3+f
#pragma unroll
          for (int o = 0; o < 8; ++o)
#pragma unroll
            for (int i = 0; i < 8; ++i)
#pragma unroll
              for (int f = 0; f < 3; ++f) {
                float r = R[o * 24 + i * 3 + f];
#pragma unroll
                for (int m = 0; m < 3; ++m) m1[o * 3 + m] += r * u[(i * 3 + f) * 3 + m];
              }
        }
      }
      __syncthreads();
    }

    // ---- scatter messages to destination nodes ----------------------------
    if (hh == 0) {
#pragma unroll
      for (int o = 0; o < 8; ++o) atomicAdd(outAcc + dN * 32 + o * 4, m0[o]);
      atomicAdd(cnt + dN, 1.0f);
    } else {
#pragma unroll
      for (int o = 0; o < 8; ++o)
#pragma unroll
        for (int m = 0; m < 3; ++m)
          atomicAdd(outAcc + dN * 32 + o * 4 + 1 + m, m1[o * 3 + m]);
    }
  }
}

// ---------------------------------------------------------------------------

extern "C" void kernel_launch(void* const* d_in, const int* in_sizes, int n_in,
                              void* d_out, int out_size, void* d_ws, size_t ws_size,
                              hipStream_t stream) {
  const int*   src    = (const int*)d_in[0];
  const int*   dst    = (const int*)d_in[1];
  const float* feat0  = (const float*)d_in[2];
  const float* feat1  = (const float*)d_in[3];
  const float* wEmb   = (const float*)d_in[4];
  const float* radial = (const float*)d_in[5];
  const float* b00    = (const float*)d_in[6];
  const float* b01    = (const float*)d_in[7];
  const float* b10    = (const float*)d_in[8];
  const float* b11    = (const float*)d_in[9];
  const float* W1     = (const float*)d_in[10];
  const float* b1     = (const float*)d_in[11];
  const float* W2     = (const float*)d_in[12];
  const float* b2     = (const float*)d_in[13];
  const float* W3_00  = (const float*)d_in[14];
  const float* b3_00  = (const float*)d_in[15];
  const float* W3_01  = (const float*)d_in[16];
  const float* b3_01  = (const float*)d_in[17];
  const float* W3_10  = (const float*)d_in[18];
  const float* b3_10  = (const float*)d_in[19];
  const float* W3_11  = (const float*)d_in[20];
  const float* b3_11  = (const float*)d_in[21];
  const float* ws0    = (const float*)d_in[22];
  const float* ws1    = (const float*)d_in[23];

  float* out = (float*)d_out;            // [N, 8, 4] flat = N*32 floats
  float* cnt = (float*)d_ws;             // N floats of scratch

  const int nz = N_NODES * 32;
  se3_zero_kernel<<<(nz + 255) / 256, 256, 0, stream>>>(out, cnt);

  se3_edge_kernel<<<NBLK, NW * 32, 0, stream>>>(
      src, dst, feat0, feat1, wEmb, radial,
      b00, b01, b10, b11,
      W1, b1, W2, b2,
      W3_00, b3_00, W3_01, b3_01, W3_10, b3_10, W3_11, b3_11,
      ws0, ws1, out, cnt);

  se3_norm_kernel<<<(nz + 255) / 256, 256, 0, stream>>>(out, cnt);
}
